// MDL_RNN_mate_25099788878090
// MI455X (gfx1250) — matrix operand order repaired and checked by reading, not compiled, not run on hardware
//
#include <hip/hip_runtime.h>
#include <hip/hip_bf16.h>

// Persistent MDL-SNN forward for MI455X (gfx1250, wave32).
//   grid = 16 WGs (one per 128-row H-chunk), block = 256 threads (8 waves).
//   Low-rank recurrent matmuls run on v_wmma_f32_16x16x32_bf16.
//   Steady state: A-fragments live in VGPRs (loaded once); B operands are kept
//   in LDS in WMMA B-fragment order (2x ds_load_b128 per fragment); the u_part
//   exchange buffer is kept in D-fragment order (2x global_store_b128 per lane).
//   d_ws layout: [0..255]   device barrier counter (memset to 0 each launch)
//                [256.. ]   u_part double buffer: 2 * 16 WGs * 64 * 16 floats (128KB)

typedef __attribute__((ext_vector_type(16))) __bf16 v16bf;
typedef __attribute__((ext_vector_type(8)))  __bf16 v8bf;
typedef __attribute__((ext_vector_type(8)))  float  v8f;

#define LM 0.90483741803596f   // exp(-DT/TAU_M), DT=1, TAU_M=10
#define LD 0.95122942450071f   // exp(-DT/TAU_D), TAU_D=20
#define LR 0.60653065971263f   // exp(-DT/TAU_R), TAU_R=2
#define VTHR 1.0f
#define TREF 2.0f

// A-fragment (16-bit, 16x32 MxK, wave32; contiguous K order used here and in bfrag_pos, not the ISA table order):
// lanes 0-15 -> M=lane, K base 0; lanes 16-31 -> M=lane-16, K base 16.
// element e: K = kbase + e  (one contiguous 32B run per lane)
__device__ inline v16bf load_a_frag(const __bf16* base, int rs, int lane) {
  const int m  = lane & 15;
  const int kb = (lane < 16) ? 0 : 16;
  v16bf a;
#pragma unroll
  for (int e = 0; e < 16; ++e) {
    const int k = kb + e;
    a[e] = base[m * rs + k];
  }
  return a;
}

// B-fragment-order LDS index for element (K, N) of a [K x 16] operand:
// tile ks = K>>5; lane = N + 16*((K>>4)&1); elem = K&15
__device__ inline int bfrag_pos(int K, int N) {
  return (K >> 5) * 512 + ((N + ((K >> 4) & 1) * 16) << 4) + (K & 15);
}

__global__ __launch_bounds__(256)
void mdl_snn_persistent(const float* __restrict__ inp,   // [T,16,4]
                        const float* __restrict__ Win,   // [H,4]
                        const float* __restrict__ Wout,  // [3,H]
                        const float* __restrict__ pin,   // [H,64]
                        const float* __restrict__ pout,  // [H,64]
                        const float* __restrict__ lvec,  // [64]
                        float* __restrict__ out,         // [T,16,3] (pre-zeroed)
                        unsigned* __restrict__ barcnt,   // device barrier counter (pre-zeroed)
                        float* __restrict__ upart,       // [2][16 wg][1024] (D-frag order)
                        int T, int H) {
  // Staging for the one-time weight load (A operands end up in registers)
  __shared__ __bf16 s_poutT[64 * 128];   // [p][h_local]
  __shared__ __bf16 s_pin  [128 * 64];   // [h_local][p]
  __shared__ __bf16 s_woutT[16 * 128];   // [o(pad16)][h_local]
  __shared__ float  s_win  [128 * 4];
  __shared__ float  s_l    [64];
  __shared__ float  s_x    [64];                     // x[t] as [b][i]
  __shared__ alignas(32) __bf16 s_rfrag[4 * 512];    // r  (128x16) in B-frag order
  __shared__ alignas(32) __bf16 s_vfrag[2 * 512];    // v  ( 64x16) in B-frag order

  const int tid   = threadIdx.x;
  const int lane  = tid & 31;
  const int wave  = tid >> 5;
  const int wg    = blockIdx.x;
  const int h0    = wg * 128;
  const int mbase = (lane >> 4) * 8;   // C/D layout: M = j + 8*(lane>=16)
  const int bcol  = lane & 15;         // C/D layout: N = lane & 15

  // ---- one-time weight load into LDS (fp32 -> bf16, zero-pad rows >= H) ----
  for (int idx = tid; idx < 64 * 128; idx += 256) {
    const int p = idx >> 7, k = idx & 127, h = h0 + k;
    s_poutT[idx] = (h < H) ? (__bf16)pout[h * 64 + p] : (__bf16)0.0f;
  }
  for (int idx = tid; idx < 128 * 64; idx += 256) {
    const int m = idx >> 6, p = idx & 63, h = h0 + m;
    s_pin[idx] = (h < H) ? (__bf16)pin[h * 64 + p] : (__bf16)0.0f;
  }
  for (int idx = tid; idx < 16 * 128; idx += 256) {
    const int m = idx >> 7, k = idx & 127, h = h0 + k;
    s_woutT[idx] = (m < 3 && h < H) ? (__bf16)Wout[m * H + h] : (__bf16)0.0f;
  }
  for (int idx = tid; idx < 128 * 4; idx += 256) {
    const int hl = idx >> 2, i = idx & 3, h = h0 + hl;
    s_win[idx] = (h < H) ? Win[h * 4 + i] : 0.0f;
  }
  if (tid < 64) s_l[tid] = lvec[tid];
  for (int idx = tid; idx < 4 * 512; idx += 256) s_rfrag[idx] = (__bf16)0.0f;
  __syncthreads();

  // ---- hoist all A-fragments into persistent VGPRs (weights are constant) ----
  v16bf aPout[4];                       // matmul1 A: M-tile = wave (waves 0-3 use it)
#pragma unroll
  for (int ks = 0; ks < 4; ++ks)
    aPout[ks] = load_a_frag(&s_poutT[((wave & 3) * 16) * 128 + ks * 32], 128, lane);
  v16bf aPin[2];                        // matmul2 A: M-tile = wave
#pragma unroll
  for (int ks = 0; ks < 2; ++ks)
    aPin[ks] = load_a_frag(&s_pin[(wave * 16) * 64 + ks * 32], 64, lane);
  v16bf aWout[4];                       // readout A (only wave 0 consumes)
#pragma unroll
  for (int ks = 0; ks < 4; ++ks)
    aWout[ks] = load_a_frag(&s_woutT[ks * 32], 128, lane);

  // ---- persistent per-lane neuron state (lane owns 8 (h,b) cells = its C/D slots) ----
  float st_mem[8], st_h[8], st_r[8], st_s[8], st_tl[8];
#pragma unroll
  for (int j = 0; j < 8; ++j) {
    st_mem[j] = 0.f; st_h[j] = 0.f; st_r[j] = 0.f; st_s[j] = 0.f; st_tl[j] = -1.f;
  }
  __syncthreads();

  for (int t = 0; t < T; ++t) {
    // ---- Phase A: partial u = poutT_chunk @ r_chunk (r from step t-1) ----
    if (wave < 4) {
      v8f acc = {};
#pragma unroll
      for (int ks = 0; ks < 4; ++ks) {
        const v16bf b = *reinterpret_cast<const v16bf*>(&s_rfrag[ks * 512 + lane * 16]);
        acc = __builtin_amdgcn_wmma_f32_16x16x32_bf16(false, aPout[ks], false, b,
                                                      (short)0, acc, false, false);
      }
      // store in D-fragment order: f = wave*256 + lane*8 + j  (32B contiguous per lane)
      float* up = upart + ((size_t)(t & 1) * 16 + wg) * 1024 + wave * 256 + lane * 8;
      float4 lo, hi;
      lo.x = acc[0]; lo.y = acc[1]; lo.z = acc[2]; lo.w = acc[3];
      hi.x = acc[4]; hi.y = acc[5]; hi.z = acc[6]; hi.w = acc[7];
      *reinterpret_cast<float4*>(up)     = lo;
      *reinterpret_cast<float4*>(up + 4) = hi;
    } else {
      const int q = tid - 128;
      if (q < 64) {
        s_x[q] = inp[t * 64 + q];                       // ((t*16+b)*4+i) = t*64 + q
        if (t + 1 < T) __builtin_prefetch(&inp[(t + 1) * 64 + q], 0, 1);
      }
    }

    // ---- device-wide step barrier (atomic counter, monotone target) ----
    __builtin_amdgcn_fence(__ATOMIC_RELEASE, "agent");
    __syncthreads();
    if (tid == 0) {
      __hip_atomic_fetch_add(barcnt, 1u, __ATOMIC_RELEASE, __HIP_MEMORY_SCOPE_AGENT);
      const unsigned tgt = 16u * (unsigned)(t + 1);
      while (__hip_atomic_load(barcnt, __ATOMIC_ACQUIRE, __HIP_MEMORY_SCOPE_AGENT) < tgt)
        __builtin_amdgcn_s_sleep(1);
    }
    __builtin_amdgcn_s_cluster_barrier();   // s_barrier_signal/wait -3 (NOP when not clustered)
    __syncthreads();
    __builtin_amdgcn_fence(__ATOMIC_ACQUIRE, "agent");

    // ---- Phase B: v = l * sum_wg u_part  (float4 loads; decode D-frag order) ----
    {
      const float* ub = upart + (size_t)(t & 1) * 16 * 1024;
      float s[4] = {0.f, 0.f, 0.f, 0.f};
#pragma unroll
      for (int w = 0; w < 16; ++w) {
        const float4 v = *reinterpret_cast<const float4*>(&ub[w * 1024 + tid * 4]);
        s[0] += v.x; s[1] += v.y; s[2] += v.z; s[3] += v.w;
      }
      // f = 4*tid + k: wave_f = f>>8, lane_f = (f>>3)&31, j_f = (f&7)
      const int f0     = tid * 4;
      const int wave_f = f0 >> 8;
      const int lane_f = (f0 >> 3) & 31;
      const int pbase  = wave_f * 16 + (f0 & 7) + ((lane_f >> 4) * 8);
      const int bf     = lane_f & 15;
#pragma unroll
      for (int k = 0; k < 4; ++k) {
        const int p = pbase + k;
        s_vfrag[bfrag_pos(p, bf)] = (__bf16)(s_l[p] * s[k]);
      }
    }
    __syncthreads();

    // ---- Phase C: I_rec = pin_chunk @ v, then LIF / double-exp update in registers ----
    {
      v8f acc = {};
#pragma unroll
      for (int ks = 0; ks < 2; ++ks) {
        const v16bf b = *reinterpret_cast<const v16bf*>(&s_vfrag[ks * 512 + lane * 16]);
        acc = __builtin_amdgcn_wmma_f32_16x16x32_bf16(false, aPin[ks], false, b,
                                                      (short)0, acc, false, false);
      }
      const float tf = (float)t;
      v8bf rnew;
#pragma unroll
      for (int j = 0; j < 8; ++j) {
        const int hl = wave * 16 + j + mbase;
        const float Iin = s_win[hl * 4 + 0] * s_x[bcol * 4 + 0]
                        + s_win[hl * 4 + 1] * s_x[bcol * 4 + 1]
                        + s_win[hl * 4 + 2] * s_x[bcol * 4 + 2]
                        + s_win[hl * 4 + 3] * s_x[bcol * 4 + 3];
        const float I    = Iin + acc[j];
        const float refr = (tf > st_tl[j] + TREF) ? 1.f : 0.f;
        const float memn = refr * (LM * st_mem[j] + (1.f - LM) * I) * (1.f - st_s[j]);
        const float hn   = LR * st_h[j] + st_s[j];
        const float rn   = LD * st_r[j] + (1.f - LD) * hn;
        const float sn   = (memn > VTHR) ? 1.f : 0.f;
        st_tl[j]  = st_tl[j] + (tf - st_tl[j]) * sn;
        st_mem[j] = memn; st_s[j] = sn; st_h[j] = hn; st_r[j] = rn;
        rnew[j] = (__bf16)rn;
      }
      // lane's 8 cells are contiguous in B-frag order: one 16B LDS store
      // pos = (wave>>1)*512 + (bcol + 16*(wave&1))*16 + mbase + j
      *reinterpret_cast<v8bf*>(
          &s_rfrag[(wave >> 1) * 512 + ((bcol + (wave & 1) * 16) << 4) + mbase]) = rnew;
    }
    __syncthreads();

    // ---- Phase D: readout y = Wout_chunk @ r_new, accumulate across WGs ----
    if (wave == 0) {
      v8f acc = {};
#pragma unroll
      for (int ks = 0; ks < 4; ++ks) {
        const v16bf b = *reinterpret_cast<const v16bf*>(&s_rfrag[ks * 512 + lane * 16]);
        acc = __builtin_amdgcn_wmma_f32_16x16x32_bf16(false, aWout[ks], false, b,
                                                      (short)0, acc, false, false);
      }
      if (lane < 16) {   // lanes 0-15 hold M=0..7 in acc[0..7]; only o=0..2 are real
        atomicAdd(&out[(t * 16 + lane) * 3 + 0], acc[0]);
        atomicAdd(&out[(t * 16 + lane) * 3 + 1], acc[1]);
        atomicAdd(&out[(t * 16 + lane) * 3 + 2], acc[2]);
      }
    }
  }
}

extern "C" void kernel_launch(void* const* d_in, const int* in_sizes, int n_in,
                              void* d_out, int out_size, void* d_ws, size_t ws_size,
                              hipStream_t stream) {
  const float* inp  = (const float*)d_in[0];  // [T,16,4,1]
  const float* Win  = (const float*)d_in[1];  // [H,4]
  const float* Wout = (const float*)d_in[2];  // [3,H]
  const float* pin  = (const float*)d_in[3];  // [H,64]
  const float* pout = (const float*)d_in[4];  // [H,64]
  const float* lv   = (const float*)d_in[5];  // [64]
  float* out = (float*)d_out;

  const int T = in_sizes[0] / 64;   // T*B*In, B=16, In=4
  const int H = in_sizes[1] / 4;    // H*In

  unsigned* cnt = (unsigned*)d_ws;
  float* upart = (float*)((char*)d_ws + 256);     // needs 2*16*64*16*4 = 128KB

  hipMemsetAsync(d_ws, 0, 256, stream);                                   // barrier counter
  hipMemsetAsync(d_out, 0, (size_t)out_size * sizeof(float), stream);     // atomicAdd target

  mdl_snn_persistent<<<dim3(16), dim3(256), 0, stream>>>(
      inp, Win, Wout, pin, pout, lv, out, cnt, upart, T, H);
}
